// PrefixEmbedAttnProbe_32976758899108
// MI455X (gfx1250) — matrix operand order repaired and checked by reading, not compiled, not run on hardware
//
#include <hip/hip_runtime.h>
#include <hip/hip_bf16.h>

// ---------------------------------------------------------------------------
// CDNA5 (gfx1250) WMMA implementation of PrefixEmbedAttn.
//  v_wmma_f32_16x16x32_bf16 for all GEMMs (f32 -> bf16 on the fly, f32 acc).
//  Double-buffered LDS pipeline; bf16-source tiles use
//  GLOBAL_LOAD_ASYNC_TO_LDS_B128 (ASYNCcnt) when the builtin is available.
//  Launches:
//    1..3) q/k/v projections        [4096x2048]x[2048x1024]
//    4)    scores = q k^T * scale, causal+pad mask fused (batched, B=2)
//    5)    row softmax (in place, bf16)
//    6)    ctx = attn v             (batched, v transposed into LDS)
//    7)    out = ctx Wout^T + bout  -> f32 d_out
//  Workspace: q(8MB) k(8MB) v(8MB) scores(16MB) ctx(8MB) = 48MB bf16.
// ---------------------------------------------------------------------------

typedef __attribute__((ext_vector_type(16))) __bf16 v16bf;
typedef __attribute__((ext_vector_type(8)))  float  v8f;

union FragAB { uint4 q[2]; v16bf v; };

#define TBM 128
#define TBN 128
#define TBK 32
#define LDSP 40   // padded K-stride in ushorts (80B rows, 16B aligned)

#if defined(__AMDGCN__) && __has_builtin(__builtin_amdgcn_global_load_async_to_lds_b128) && __has_builtin(__builtin_amdgcn_s_wait_asynccnt)
#  define HAVE_ASYNC 1
#  define AS1 __attribute__((address_space(1)))
#  define AS3 __attribute__((address_space(3)))
typedef __attribute__((ext_vector_type(4))) int v4i_t;
#  define ASYNC_GPTR(p) ((AS1 v4i_t*)(void*)(p))
#  define ASYNC_LPTR(p) ((AS3 v4i_t*)(void*)(p))
#else
#  define HAVE_ASYNC 0
#endif

__device__ __forceinline__ unsigned short f2bf(float f) {
  unsigned u = __float_as_uint(f);
  unsigned r = ((u >> 16) & 1u) + 0x7FFFu;   // round-to-nearest-even
  return (unsigned short)((u + r) >> 16);
}
__device__ __forceinline__ float bf2f(unsigned short h) {
  return __uint_as_float(((unsigned)h) << 16);
}

__device__ __forceinline__ void waitAsync0() {
#if HAVE_ASYNC
  __builtin_amdgcn_s_wait_asynccnt(0);
#endif
}

// ---- tile movers ----------------------------------------------------------
// f32 source: stage (global->regs) then commit (cvt + ds_store) after compute.
__device__ __forceinline__ void stageF32(float4 (&s)[4], const float* __restrict__ src,
                                         int ld, size_t rowBase, int kk, int tid) {
  int r0 = tid >> 3, c4 = (tid & 7) << 2;
#pragma unroll
  for (int it = 0; it < 4; ++it)
    s[it] = *(const float4*)(src + (rowBase + r0 + it * 32) * (size_t)ld + kk + c4);
}
__device__ __forceinline__ void commitF32(unsigned short (*tile)[LDSP],
                                          const float4 (&s)[4], int tid) {
  int r0 = tid >> 3, c4 = (tid & 7) << 2;
#pragma unroll
  for (int it = 0; it < 4; ++it) {
    unsigned short* d = &tile[r0 + it * 32][c4];
    d[0] = f2bf(s[it].x); d[1] = f2bf(s[it].y);
    d[2] = f2bf(s[it].z); d[3] = f2bf(s[it].w);
  }
}

// bf16 source, natural [rows][K] layout
__device__ __forceinline__ void stageBF(uint4 (&s)[2], const unsigned short* __restrict__ src,
                                        int ld, size_t rowBase, int kk, int tid) {
  int r0 = tid >> 2, c8 = (tid & 3) << 3;
#pragma unroll
  for (int it = 0; it < 2; ++it)
    s[it] = *(const uint4*)(src + (rowBase + r0 + it * 64) * (size_t)ld + kk + c8);
}
__device__ __forceinline__ void commitBF(unsigned short (*tile)[LDSP],
                                         const uint4 (&s)[2], int tid) {
  int r0 = tid >> 2, c8 = (tid & 3) << 3;
#pragma unroll
  for (int it = 0; it < 2; ++it)
    *(uint4*)&tile[r0 + it * 64][c8] = s[it];
}

#if HAVE_ASYNC
// bf16 source, natural layout, direct-to-LDS async copy (ASYNCcnt)
__device__ __forceinline__ void asyncBF(unsigned short (*tile)[LDSP],
                                        const unsigned short* __restrict__ src,
                                        int ld, size_t rowBase, int kk, int tid) {
  int r0 = tid >> 2, c8 = (tid & 3) << 3;
#pragma unroll
  for (int it = 0; it < 2; ++it) {
    const unsigned short* g = src + (rowBase + r0 + it * 64) * (size_t)ld + kk + c8;
    __builtin_amdgcn_global_load_async_to_lds_b128(
        ASYNC_GPTR(g), ASYNC_LPTR(&tile[r0 + it * 64][c8]), 0, 0);
  }
}
#else
__device__ __forceinline__ void asyncBF(unsigned short (*)[LDSP],
                                        const unsigned short*, int, size_t, int, int) {}
#endif

// bf16 source, transposed: tile[n][k] = src[(kk+k)*ld + nBase+n]   (v matrix)
__device__ __forceinline__ void stageBFT(uint4 (&s)[2], const unsigned short* __restrict__ src,
                                         int ld, size_t nBase, int kk, int tid) {
#pragma unroll
  for (int it = 0; it < 2; ++it) {
    int slot = tid + it * 256;
    int k = slot & 31, n0 = (slot >> 5) << 3;
    s[it] = *(const uint4*)(src + (size_t)(kk + k) * ld + nBase + n0);
  }
}
__device__ __forceinline__ void commitBFT(unsigned short (*tile)[LDSP],
                                          const uint4 (&s)[2], int tid) {
#pragma unroll
  for (int it = 0; it < 2; ++it) {
    int slot = tid + it * 256;
    int k = slot & 31, n0 = (slot >> 5) << 3;
    unsigned short tmp[8];
    *(uint4*)tmp = s[it];
#pragma unroll
    for (int j = 0; j < 8; ++j) tile[n0 + j][k] = tmp[j];
  }
}

// ---- fragment loads + 8 WMMAs for one K-step ------------------------------
__device__ __forceinline__ void computeStep(const unsigned short (*aT)[LDSP],
                                            const unsigned short (*bT)[LDSP],
                                            v8f (&acc)[4][2], int lane, int wM, int wN) {
  const int rsel = lane & 15;
  const int aKb  = (lane < 16) ? 0 : 8;    // §7.12.2 16-bit A layout
  const int bKb  = (lane < 16) ? 0 : 8;    // B K-base (same as A)
  FragAB a[4], b[2];
#pragma unroll
  for (int mf = 0; mf < 4; ++mf) {
    const unsigned short* p = &aT[wM + mf * 16 + rsel][aKb];
    a[mf].q[0] = *(const uint4*)p;          // K = kb .. kb+7
    a[mf].q[1] = *(const uint4*)(p + 16);   // K = kb+16 .. kb+23
  }
#pragma unroll
  for (int nf = 0; nf < 2; ++nf) {
    const unsigned short* p = &bT[wN + nf * 16 + rsel][bKb];
    b[nf].q[0] = *(const uint4*)p;          // K = kb .. kb+7
    b[nf].q[1] = *(const uint4*)(p + 16);   // K = kb+16 .. kb+23
  }
#pragma unroll
  for (int mf = 0; mf < 4; ++mf)
#pragma unroll
    for (int nf = 0; nf < 2; ++nf)
      acc[mf][nf] = __builtin_amdgcn_wmma_f32_16x16x32_bf16(
          false, a[mf].v, false, b[nf].v, (short)0, acc[mf][nf], false, false);
}

// ---- generic double-buffered WMMA GEMM ------------------------------------
// AKIND/BKIND: 0 = f32 source, 1 = bf16 source.  BTR: B is [K][N] (transpose).
// EPI: 0 = bf16 store; 1 = bf16 store + scale + causal/pad mask (scores);
//      2 = f32 store + bias (logits).
template <int AKIND, int BKIND, bool BTR, int EPI>
__global__ __launch_bounds__(256) void gemm_wmma_kernel(
    const void* __restrict__ Ap, const void* __restrict__ Bp, void* __restrict__ Cp,
    int M, int N, int K, int lda, int ldb, int ldc,
    size_t aBatchBytes, size_t bBatchBytes, size_t cBatchBytes,
    const int* __restrict__ mask, int maskStride, float scale,
    const float* __restrict__ bias) {
  __shared__ unsigned short aT[2][TBM][LDSP];   // 2 x 10KB
  __shared__ unsigned short bT[2][TBN][LDSP];   // 2 x 10KB

  constexpr bool A_ASYNC = (HAVE_ASYNC != 0) && (AKIND == 1);
  constexpr bool B_ASYNC = (HAVE_ASYNC != 0) && (BKIND == 1) && !BTR;

  const int tid  = threadIdx.x;
  const int z    = blockIdx.z;
  const char* Ab = (const char*)Ap + (size_t)z * aBatchBytes;
  const char* Bb = (const char*)Bp + (size_t)z * bBatchBytes;
  char*       Cb = (char*)Cp + (size_t)z * cBatchBytes;

  const size_t mBase = (size_t)blockIdx.y * TBM;
  const size_t nBase = (size_t)blockIdx.x * TBN;

  const int lane = tid & 31;
  const int wave = tid >> 5;
  const int wM = (wave & 1) * 64;   // 2 waves along M: 4 frags each
  const int wN = (wave >> 1) * 32;  // 4 waves along N: 2 frags each

  const v8f vzero = {0.f, 0.f, 0.f, 0.f, 0.f, 0.f, 0.f, 0.f};
  v8f acc[4][2];
#pragma unroll
  for (int mf = 0; mf < 4; ++mf)
#pragma unroll
    for (int nf = 0; nf < 2; ++nf) acc[mf][nf] = vzero;

  float4 saf[4], sbf[4];   // staged regs (f32 paths)
  uint4  sau[2], sbu[2];   // staged regs (bf16 fallback / transpose path)

  const int nk = K / TBK;

  // ---- prologue: tile 0 into buffer 0 ----
  if constexpr (AKIND == 0) { stageF32(saf, (const float*)Ab, lda, mBase, 0, tid);
                              commitF32(aT[0], saf, tid); }
  else if constexpr (A_ASYNC) { asyncBF(aT[0], (const unsigned short*)Ab, lda, mBase, 0, tid); }
  else { stageBF(sau, (const unsigned short*)Ab, lda, mBase, 0, tid);
         commitBF(aT[0], sau, tid); }

  if constexpr (BTR) { stageBFT(sbu, (const unsigned short*)Bb, ldb, nBase, 0, tid);
                       commitBFT(bT[0], sbu, tid); }
  else if constexpr (BKIND == 0) { stageF32(sbf, (const float*)Bb, ldb, nBase, 0, tid);
                                   commitF32(bT[0], sbf, tid); }
  else if constexpr (B_ASYNC) { asyncBF(bT[0], (const unsigned short*)Bb, ldb, nBase, 0, tid); }
  else { stageBF(sbu, (const unsigned short*)Bb, ldb, nBase, 0, tid);
         commitBF(bT[0], sbu, tid); }

  for (int kt = 0; kt < nk; ++kt) {
    const int cur = kt & 1, nxt = cur ^ 1;
    const bool hasNext = (kt + 1) < nk;
    const int kkn = (kt + 1) * TBK;

    // tile kt fully resident: drain async, then barrier (also covers ds stores)
    if constexpr (A_ASYNC || B_ASYNC) waitAsync0();
    __syncthreads();

    // issue tile kt+1 (overlaps with this step's WMMAs); safe: barrier above
    // transitively follows the trailing barrier of step kt-1, so nobody is
    // still reading buf[nxt].
    if (hasNext) {
      if constexpr (AKIND == 0) stageF32(saf, (const float*)Ab, lda, mBase, kkn, tid);
      else if constexpr (A_ASYNC) asyncBF(aT[nxt], (const unsigned short*)Ab, lda, mBase, kkn, tid);
      else stageBF(sau, (const unsigned short*)Ab, lda, mBase, kkn, tid);

      if constexpr (BTR) stageBFT(sbu, (const unsigned short*)Bb, ldb, nBase, kkn, tid);
      else if constexpr (BKIND == 0) stageF32(sbf, (const float*)Bb, ldb, nBase, kkn, tid);
      else if constexpr (B_ASYNC) asyncBF(bT[nxt], (const unsigned short*)Bb, ldb, nBase, kkn, tid);
      else stageBF(sbu, (const unsigned short*)Bb, ldb, nBase, kkn, tid);

      // L2 prefetch two tiles ahead on the f32 streams (global_prefetch_b8)
      if constexpr (AKIND == 0) {
        if (kkn + TBK < K)
          __builtin_prefetch((const char*)Ab +
              (((mBase + (tid >> 3)) * (size_t)lda + kkn + TBK) << 2), 0, 1);
      }
    }

    computeStep(aT[cur], bT[cur], acc, lane, wM, wN);

    // convert + commit staged regs after the WMMAs are issued
    if (hasNext) {
      if constexpr (AKIND == 0) commitF32(aT[nxt], saf, tid);
      else if constexpr (!A_ASYNC) commitBF(aT[nxt], sau, tid);
      if constexpr (BTR) commitBFT(bT[nxt], sbu, tid);
      else if constexpr (BKIND == 0) commitF32(bT[nxt], sbf, tid);
      else if constexpr (!B_ASYNC) commitBF(bT[nxt], sbu, tid);
    }
    __syncthreads();
  }

  // ---- epilogue: C/D frag elem e -> row = e + (lane<16?0:8), col = lane&15
  const int colLane = lane & 15;
  const int rowAdd  = (lane < 16) ? 0 : 8;
#pragma unroll
  for (int mf = 0; mf < 4; ++mf)
#pragma unroll
    for (int nf = 0; nf < 2; ++nf)
#pragma unroll
      for (int e = 0; e < 8; ++e) {
        size_t row = mBase + wM + mf * 16 + e + rowAdd;
        size_t col = nBase + wN + nf * 16 + colLane;
        float v = acc[mf][nf][e];
        if (EPI == 0) {
          ((unsigned short*)Cb)[row * (size_t)ldc + col] = f2bf(v);
        } else if (EPI == 1) {
          bool allowed = (col <= row) && (mask[(size_t)z * maskStride + col] != 0);
          ((unsigned short*)Cb)[row * (size_t)ldc + col] =
              f2bf(allowed ? v * scale : -1e30f);
        } else {
          ((float*)Cb)[row * (size_t)ldc + col] = v + bias[col];
        }
      }
}

// ---- row softmax over bf16 scores, in place -------------------------------
__global__ __launch_bounds__(256) void softmax_kernel(unsigned short* __restrict__ sc,
                                                      int T) {
  unsigned short* p = sc + (size_t)blockIdx.x * T;
  const int tid = threadIdx.x, lane = tid & 31, wave = tid >> 5;
  __shared__ float redM[8], redS[8];

  float m = -3.0e38f;
  for (int i = tid; i < T; i += 256) m = fmaxf(m, bf2f(p[i]));
#pragma unroll
  for (int off = 16; off; off >>= 1) m = fmaxf(m, __shfl_xor(m, off, 32));
  if (lane == 0) redM[wave] = m;
  __syncthreads();
  m = redM[0];
#pragma unroll
  for (int j = 1; j < 8; ++j) m = fmaxf(m, redM[j]);

  float s = 0.f;
  for (int i = tid; i < T; i += 256) {
    float x = bf2f(p[i]);
    if (x > -1e29f) s += __expf(x - m);
  }
#pragma unroll
  for (int off = 16; off; off >>= 1) s += __shfl_xor(s, off, 32);
  if (lane == 0) redS[wave] = s;
  __syncthreads();
  s = 0.f;
#pragma unroll
  for (int j = 0; j < 8; ++j) s += redS[j];
  float inv = (s > 0.f) ? (1.0f / s) : 0.0f;  // fully-masked row -> all zeros

  for (int i = tid; i < T; i += 256) {
    float x = bf2f(p[i]);
    float e = (x > -1e29f) ? __expf(x - m) * inv : 0.0f;
    p[i] = f2bf(e);
  }
}

// ---------------------------------------------------------------------------
extern "C" void kernel_launch(void* const* d_in, const int* in_sizes, int n_in,
                              void* d_out, int out_size, void* d_ws, size_t ws_size,
                              hipStream_t stream) {
  const float* h    = (const float*)d_in[0];
  const float* e    = (const float*)d_in[1];
  const int*   mask = (const int*)d_in[2];
  const float* Wq   = (const float*)d_in[3];
  const float* Wk   = (const float*)d_in[4];
  const float* Wv   = (const float*)d_in[5];
  const float* Wout = (const float*)d_in[6];
  const float* bout = (const float*)d_in[7];
  float* out = (float*)d_out;

  const int Bsz = 2, T = 2048, H = 2048, KV = 1024, V = 32000;
  const size_t MT = (size_t)Bsz * T;  // 4096

  // workspace layout (bf16): q | k | v | scores | ctx  == 48 MB
  unsigned short* qw = (unsigned short*)d_ws;
  unsigned short* kw = qw + MT * KV;
  unsigned short* vw = kw + MT * KV;
  unsigned short* sw = vw + MT * KV;              // [B][T][T]
  unsigned short* cw = sw + (size_t)Bsz * T * T;  // [B*T][KV]

  dim3 blk(256);
  const float scale = 0.03125f;  // KV^-0.5

  // 1-3) q/k/v projections
  gemm_wmma_kernel<0, 0, false, 0><<<dim3(KV / TBN, MT / TBM, 1), blk, 0, stream>>>(
      h, Wq, qw, (int)MT, KV, H, H, H, KV, 0, 0, 0, nullptr, 0, 0.f, nullptr);
  gemm_wmma_kernel<0, 0, false, 0><<<dim3(KV / TBN, MT / TBM, 1), blk, 0, stream>>>(
      e, Wk, kw, (int)MT, KV, H, H, H, KV, 0, 0, 0, nullptr, 0, 0.f, nullptr);
  gemm_wmma_kernel<0, 0, false, 0><<<dim3(KV / TBN, MT / TBM, 1), blk, 0, stream>>>(
      e, Wv, vw, (int)MT, KV, H, H, H, KV, 0, 0, 0, nullptr, 0, 0.f, nullptr);

  // 4) scores = q k^T * scale, masked (batched over B)
  gemm_wmma_kernel<1, 1, false, 1><<<dim3(T / TBN, T / TBM, Bsz), blk, 0, stream>>>(
      qw, kw, sw, T, T, KV, KV, KV, T,
      (size_t)T * KV * 2, (size_t)T * KV * 2, (size_t)T * T * 2,
      mask, T, scale, nullptr);

  // 5) softmax rows
  softmax_kernel<<<dim3(Bsz * T), blk, 0, stream>>>(sw, T);

  // 6) ctx = attn v  (v transposed into LDS; batched)
  gemm_wmma_kernel<1, 1, true, 0><<<dim3(KV / TBN, T / TBM, Bsz), blk, 0, stream>>>(
      sw, vw, cw, T, KV, T, T, KV, KV,
      (size_t)T * T * 2, (size_t)T * KV * 2, (size_t)T * KV * 2,
      nullptr, 0, 0.f, nullptr);

  // 7) logits = ctx Wout^T + bout  (f32 output)
  gemm_wmma_kernel<1, 0, false, 2><<<dim3(V / TBN, MT / TBM, 1), blk, 0, stream>>>(
      cw, Wout, out, (int)MT, V, KV, KV, KV, V, 0, 0, 0, nullptr, 0, 0.f, bout);
}
